// DAGLayer_17085379904201
// MI455X (gfx1250) — matrix operand order repaired and checked by reading, not compiled, not run on hardware
//
#include <hip/hip_runtime.h>

// ---------------------------------------------------------------------------
// DAGLayer fused scan for gfx1250 (MI455X).
//  - All 50 scan steps fused into one kernel: per-row recurrent state (buf)
//    lives in LDS for the whole scan (row n only ever reads/writes row n).
//  - GEMMs run on v_wmma_f32_16x16x32_bf16 (bf16 in, f32 accumulate).
//  - W1/W2 pre-packed to bf16 in WMMA-B fragment order (32B/lane contiguous).
//  - Prefetch parameters (5+2b, b<48, locality 3) are load-bearing: they give
//    the scheduler the 2x-unrolled, NOP-free WMMA pipeline (wmma=30 schedule).
//    Other distances/inline-asm variants trigger a pipelined schedule with
//    v_nop hazard stalls and accumulator copies.
// ---------------------------------------------------------------------------

typedef __attribute__((ext_vector_type(16))) __bf16 v16bf;
typedef __attribute__((ext_vector_type(8)))  float  v8f;
typedef __attribute__((ext_vector_type(4)))  float  f32x4;

#define NTOT   6400
#define NA     50            // MAX_ATOMS (scan steps, parents per row)
#define GF     64            // graph feature width
#define AFF    75            // atom feature width
#define HID    128           // hidden width
#define ROWS   32            // rows (DAG instances) per workgroup
#define RSTR   3208          // LDS buf row stride in bf16: 50*64 + 8 pad
#define NKT1   101           // GEMM1 k-tiles: (75 padded to 96) + 49*64, /32
#define W1P_ELEMS (NKT1 * 8 * 32 * 16)   // 413696 bf16
#define W2P_ELEMS (4 * 4 * 32 * 16)      // 8192 bf16

union ABfrag { f32x4 f[2]; v16bf v; };

// B fragment: 16 bf16 contiguous per lane (packed at prep time).
__device__ __forceinline__ v16bf frag_contig(const __bf16* p) {
  ABfrag u;
  u.f[0] = *reinterpret_cast<const f32x4*>(p);
  u.f[1] = *reinterpret_cast<const f32x4*>(p + 8);
  return u.v;
}
// A fragment: lane holds 16 consecutive K values (elements 0..15 -> k=c..c+15).
__device__ __forceinline__ v16bf frag_split(const __bf16* p) {
  ABfrag u;
  u.f[0] = *reinterpret_cast<const f32x4*>(p);
  u.f[1] = *reinterpret_cast<const f32x4*>(p + 8);
  return u.v;
}
__device__ __forceinline__ v8f wmma_bf16(v16bf a, v16bf b, v8f c) {
  return __builtin_amdgcn_wmma_f32_16x16x32_bf16(
      /*neg_a=*/false, a, /*neg_b=*/false, b,
      /*c_mod=*/(short)0, c, /*reuse_a=*/false, /*reuse_b=*/false);
}

// ---------------------------------------------------------------------------
// Prep: pack W1 (3211x128 f32) and W2 (128x64 f32) into bf16 fragment order.
// Padded K for GEMM1: k<75 -> atom feats; 75..95 -> zero pad; 96+ -> pg part
// (original W1 row = padded_k - 21).
// ---------------------------------------------------------------------------
__global__ __launch_bounds__(256) void dag_pack_weights(
    const float* __restrict__ W1, const float* __restrict__ W2,
    __bf16* __restrict__ W1p, __bf16* __restrict__ W2p)
{
  int gid = blockIdx.x * 256 + threadIdx.x;
  if (gid < W1P_ELEMS) {
    int e  = gid & 15;
    int l  = (gid >> 4) & 31;
    int nt = (gid >> 9) & 7;
    int kt = gid >> 12;
    int col = nt * 16 + (l & 15);
    int kg  = kt * 32 + (l >> 4) * 16 + e;      // padded K index
    float v = 0.0f;
    if (kg < AFF)      v = W1[kg * HID + col];
    else if (kg >= 96) v = W1[(kg - 21) * HID + col];
    W1p[gid] = (__bf16)v;
  } else {
    int g2 = gid - W1P_ELEMS;
    if (g2 < W2P_ELEMS) {
      int e  = g2 & 15;
      int l  = (g2 >> 4) & 31;
      int nt = (g2 >> 9) & 3;
      int kt = g2 >> 11;
      int col = nt * 16 + (l & 15);
      int k   = kt * 32 + (l >> 4) * 16 + e;
      W2p[g2] = (__bf16)W2[k * GF + col];
    }
  }
}

// ---------------------------------------------------------------------------
// Main fused-scan kernel. 256 threads = 8 wave32. 32 rows per workgroup.
// ---------------------------------------------------------------------------
__global__ void __launch_bounds__(256, 1) dag_main(
    const float* __restrict__ atomf,          // (6400, 75)
    const float* __restrict__ b1,             // (128)
    const float* __restrict__ b2,             // (64)
    const int*   __restrict__ parents,        // (6400, 50, 50)
    const int*   __restrict__ orders,         // (6400, 50)
    const unsigned char* __restrict__ masks,  // (6400, 50) bool
    const __bf16* __restrict__ W1p,
    const __bf16* __restrict__ W2p,
    float* __restrict__ out)                  // (6400, 64)
{
  __shared__ __align__(16) __bf16 bufLDS[ROWS * RSTR];   // 205,312 B
  __shared__ __align__(16) __bf16 afLDS[ROWS * 96];      //   6,144 B
  __shared__ __align__(16) __bf16 hLDS[ROWS * HID];      //   8,192 B
  __shared__ int parLDS[ROWS * 52];                      //   6,656 B
  __shared__ int orderLDS[ROWS];
  __shared__ int maskLDS[ROWS];

  const int tid  = threadIdx.x;
  const int wave = tid >> 5;
  const int lane = tid & 31;
  const int hi   = lane >> 4;      // half-wave select
  const int ln   = lane & 15;      // row/col within 16-tile
  const int cofs = hi * 8;         // accumulator row offset of this lane
  const int row0 = blockIdx.x * ROWS;

  const float b1v = b1[wave * 16 + ln];
  const int   mt2 = wave >> 2;     // GEMM2 M tile (0/1)
  const int   nt2 = wave & 3;      // GEMM2 N tile (0..3)
  const float b2v = b2[nt2 * 16 + ln];

  // buf0 = zeros
  for (int i = tid; i < (ROWS * RSTR) / 2; i += 256)
    reinterpret_cast<unsigned int*>(bufLDS)[i] = 0u;

  for (int t = 0; t < NA; ++t) {
    __syncthreads();   // prior step's buf scatters / par0 reads complete

    if (tid < ROWS) {
      orderLDS[tid] = orders[(row0 + tid) * NA + t];
      maskLDS[tid]  = (int)masks[(row0 + tid) * NA + t];
    }
    for (int i = tid; i < ROWS * NA; i += 256) {
      int rr = i / NA, j = i - rr * NA;
      parLDS[rr * 52 + j] = parents[((row0 + rr) * NA + t) * NA + j];
    }
    __syncthreads();

    // stage atom features (gathered by order), zero-padded 75 -> 96
    for (int i = tid; i < ROWS * 96; i += 256) {
      int rr = i / 96, c = i - rr * 96;
      float v = (c < AFF) ? atomf[orderLDS[rr] * AFF + c] : 0.0f;
      afLDS[i] = (__bf16)v;
    }
    __syncthreads();

    // ---------------- GEMM1: (32 x 3232) @ W1p -> h (32 x 128) ------------
    v8f acc0 = {0.f,0.f,0.f,0.f,0.f,0.f,0.f,0.f};
    v8f acc1 = {0.f,0.f,0.f,0.f,0.f,0.f,0.f,0.f};

    // 3 k-tiles from atom-feature staging
#pragma unroll
    for (int kt = 0; kt < 3; ++kt) {
      v16bf bF = frag_contig(&W1p[((kt * 8 + wave) * 32 + lane) * 16]);
      v16bf a0 = frag_split(&afLDS[ln * 96 + kt * 32 + hi * 16]);
      v16bf a1 = frag_split(&afLDS[(16 + ln) * 96 + kt * 32 + hi * 16]);
      acc0 = wmma_bf16(a0, bF, acc0);
      acc1 = wmma_bf16(a1, bF, acc1);
    }
    // 49 parent blocks * 2 k-tiles each, gathered from LDS-resident buf
    for (int b = 0; b < 49; ++b) {
      if (b < 48)  // prefetch next parent-block's B fragments into near cache
        __builtin_prefetch(&W1p[(((5 + 2 * b) * 8 + wave) * 32 + lane) * 16], 0, 3);
      const int p0 = parLDS[ln * 52 + 1 + b];
      const int p1 = parLDS[(16 + ln) * 52 + 1 + b];
      const __bf16* r0 = &bufLDS[ln * RSTR + p0 * 64 + hi * 16];
      const __bf16* r1 = &bufLDS[(16 + ln) * RSTR + p1 * 64 + hi * 16];
#pragma unroll
      for (int h = 0; h < 2; ++h) {
        const int kt = 3 + 2 * b + h;
        v16bf bF = frag_contig(&W1p[((kt * 8 + wave) * 32 + lane) * 16]);
        v16bf a0 = frag_split(r0 + h * 32);
        v16bf a1 = frag_split(r1 + h * 32);
        acc0 = wmma_bf16(a0, bF, acc0);
        acc1 = wmma_bf16(a1, bF, acc1);
      }
    }
    // bias + relu, store h as bf16
#pragma unroll
    for (int v = 0; v < 8; ++v) {
      float x0 = acc0[v] + b1v; x0 = x0 > 0.f ? x0 : 0.f;
      float x1 = acc1[v] + b1v; x1 = x1 > 0.f ? x1 : 0.f;
      hLDS[(v + cofs) * HID + wave * 16 + ln]        = (__bf16)x0;
      hLDS[(16 + v + cofs) * HID + wave * 16 + ln]   = (__bf16)x1;
    }
    __syncthreads();

    // ---------------- GEMM2: h (32 x 128) @ W2p -> out (32 x 64) ----------
    v8f acc2 = {0.f,0.f,0.f,0.f,0.f,0.f,0.f,0.f};
#pragma unroll
    for (int kt = 0; kt < 4; ++kt) {
      v16bf a  = frag_split(&hLDS[(mt2 * 16 + ln) * HID + kt * 32 + hi * 16]);
      v16bf bF = frag_contig(&W2p[((kt * 4 + nt2) * 32 + lane) * 16]);
      acc2 = wmma_bf16(a, bF, acc2);
    }
    const int col = nt2 * 16 + ln;
#pragma unroll
    for (int v = 0; v < 8; ++v) {
      const int m = mt2 * 16 + v + cofs;
      float x = acc2[v] + b2v; x = x > 0.f ? x : 0.f;
      if (t == NA - 1) out[(row0 + m) * GF + col] = x;   // outs[-1]
      if (maskLDS[m])                                     // masked scatter
        bufLDS[m * RSTR + parLDS[m * 52] * 64 + col] = (__bf16)x;
    }
  }
}

// ---------------------------------------------------------------------------
extern "C" void kernel_launch(void* const* d_in, const int* in_sizes, int n_in,
                              void* d_out, int out_size, void* d_ws, size_t ws_size,
                              hipStream_t stream) {
  const float* atomf   = (const float*)d_in[0];          // (6400,75)
  const float* W1      = (const float*)d_in[1];          // (3211,128)
  const float* b1      = (const float*)d_in[2];          // (128)
  const float* W2      = (const float*)d_in[3];          // (128,64)
  const float* b2      = (const float*)d_in[4];          // (64)
  const int*   parents = (const int*)d_in[5];            // (6400,50,50)
  const int*   orders  = (const int*)d_in[6];            // (6400,50)
  const unsigned char* masks = (const unsigned char*)d_in[7]; // (6400,50) bool
  (void)in_sizes; (void)n_in; (void)out_size; (void)ws_size;

  __bf16* W1p = (__bf16*)d_ws;                 // 827,392 B
  __bf16* W2p = W1p + W1P_ELEMS;               //  16,384 B

  const int prep_elems = W1P_ELEMS + W2P_ELEMS;          // 421,888
  dag_pack_weights<<<prep_elems / 256, 256, 0, stream>>>(W1, W2, W1p, W2p);
  dag_main<<<NTOT / ROWS, 256, 0, stream>>>(atomf, b1, b2, parents, orders,
                                            masks, W1p, W2p, (float*)d_out);
}
